// MoERagged_16441134809276
// MI455X (gfx1250) — matrix operand order repaired and checked by reading, not compiled, not run on hardware
//
#include <hip/hip_runtime.h>
#include <math.h>
#include <stdint.h>

// ---- Problem constants (from reference setup_inputs) ----
#define GDIM 4
#define SDIM 2048
#define FDIM 2048
#define HDIM 4096
#define EDIM 8
#define TOPK 2
#define BTOK (GDIM * SDIM)          // 8192 tokens
#define RROWS (BTOK * TOPK)         // 16384 routed rows
#define MAXTILES (RROWS / 16 + EDIM) // 1032 worst-case 16-row tiles

typedef __attribute__((ext_vector_type(16))) _Float16 v16h;
typedef __attribute__((ext_vector_type(8)))  float    v8f;

__device__ __forceinline__ float gelu_tanh(float v) {
  const float c = 0.7978845608028654f;
  float inner = c * (v + 0.044715f * v * v * v);
  return 0.5f * v * (1.0f + tanhf(inner));
}

// Low 32 bits of a generic pointer to LDS == LDS byte offset (ISA 10.2:
// LDS aperture addresses map addr[31:0] to the wave's LDS space).
__device__ __forceinline__ unsigned lds_off32(const void* p) {
  return (unsigned)(uintptr_t)p;
}

// Async DMA: global -> LDS, 16 bytes per lane, tracked by ASYNCcnt.
__device__ __forceinline__ void async_copy_b128(unsigned lds_addr, const void* gaddr) {
  asm volatile("global_load_async_to_lds_b128 %0, %1, off"
               :
               : "v"(lds_addr), "v"((uint64_t)(uintptr_t)gaddr)
               : "memory");
}

// Wait until at most N async (global<->LDS DMA) ops remain outstanding.
// Async loads complete in order, so N==8 means "previous group of 8 landed".
template <int N>
__device__ __forceinline__ void wait_asynccnt() {
  asm volatile("s_wait_asynccnt %0" ::"i"(N) : "memory");
}

// ---------------- init: zero counters ----------------
__global__ void init_kernel(int* counts, int* cursor, int* numTiles) {
  int tid = threadIdx.x;
  if (tid < EDIM) { counts[tid] = 0; cursor[tid] = 0; }
  if (tid == 0) *numTiles = 0;
}

// ---------------- router: RMSnorm + logits + softmax + top2 ----------------
// One wave32 per token. per_expert_scale folded into combine weight.
__global__ void router_kernel(const float* __restrict__ x,
                              const float* __restrict__ w_router,
                              const float* __restrict__ pes,
                              const float* __restrict__ rscale,
                              int* __restrict__ expertOf,
                              float* __restrict__ wgt,
                              int* __restrict__ counts) {
  int lane = threadIdx.x & 31;
  int wave = threadIdx.x >> 5;
  int t = blockIdx.x * 8 + wave;
  if (t >= BTOK) return;
  const float* xr = x + (size_t)t * FDIM;
  float ss = 0.f;
  float acc[EDIM];
#pragma unroll
  for (int e = 0; e < EDIM; ++e) acc[e] = 0.f;
  for (int f = lane; f < FDIM; f += 32) {
    float xv = xr[f];
    ss = fmaf(xv, xv, ss);
    float xs = xv * rscale[f];
    const float* wr = w_router + (size_t)f * EDIM;
#pragma unroll
    for (int e = 0; e < EDIM; ++e) acc[e] = fmaf(xs, wr[e], acc[e]);
  }
#pragma unroll
  for (int off = 16; off > 0; off >>= 1) {
    ss += __shfl_xor(ss, off, 32);
#pragma unroll
    for (int e = 0; e < EDIM; ++e) acc[e] += __shfl_xor(acc[e], off, 32);
  }
  if (lane == 0) {
    float rstd = rsqrtf(ss * (1.0f / FDIM) + 1e-6f);
    float sc = rstd * rsqrtf((float)FDIM);
    float lg[EDIM], mx = -1e30f;
#pragma unroll
    for (int e = 0; e < EDIM; ++e) { lg[e] = acc[e] * sc; mx = fmaxf(mx, lg[e]); }
    float sum = 0.f, p[EDIM];
#pragma unroll
    for (int e = 0; e < EDIM; ++e) { p[e] = __expf(lg[e] - mx); sum += p[e]; }
    float inv = 1.0f / sum;
#pragma unroll
    for (int e = 0; e < EDIM; ++e) p[e] *= inv;
    int c0 = 0;
#pragma unroll
    for (int e = 1; e < EDIM; ++e) if (lg[e] > lg[c0]) c0 = e;
    int c1 = (c0 == 0) ? 1 : 0;
#pragma unroll
    for (int e = 0; e < EDIM; ++e) if (e != c0 && lg[e] > lg[c1]) c1 = e;
    float rn = p[c0] + p[c1];
    rn = (rn > 0.f) ? rn : 1.f;
    expertOf[t * 2 + 0] = c0;
    expertOf[t * 2 + 1] = c1;
    wgt[t * 2 + 0] = p[c0] / rn * pes[c0];
    wgt[t * 2 + 1] = p[c1] / rn * pes[c1];
    atomicAdd(&counts[c0], 1);
    atomicAdd(&counts[c1], 1);
  }
}

// ---------------- plan: offsets + expert-major tile list ----------------
__global__ void plan_kernel(const int* __restrict__ counts, int* __restrict__ offsets,
                            int* __restrict__ tileE, int* __restrict__ tileR0,
                            int* __restrict__ tileRows, int* __restrict__ numTiles) {
  if (threadIdx.x != 0 || blockIdx.x != 0) return;
  int off = 0;
  for (int e = 0; e < EDIM; ++e) { offsets[e] = off; off += counts[e]; }
  offsets[EDIM] = off;
  int nt = 0;
  for (int e = 0; e < EDIM; ++e) {
    int beg = offsets[e], end = beg + counts[e];
    for (int r = beg; r < end; r += 16) {
      tileE[nt] = e;
      tileR0[nt] = r;
      tileRows[nt] = min(16, end - r);
      ++nt;
    }
  }
  *numTiles = nt;
}

// ---------------- scatter: bucket rows by expert ----------------
__global__ void scatter_kernel(const int* __restrict__ expertOf, const float* __restrict__ wgt,
                               const int* __restrict__ offsets, int* __restrict__ cursor,
                               int* __restrict__ sTok, int* __restrict__ sSlot,
                               float* __restrict__ sWgt) {
  int r = blockIdx.x * blockDim.x + threadIdx.x;
  if (r >= RROWS) return;
  int e = expertOf[r];
  int pos = atomicAdd(&cursor[e], 1);
  int idx = offsets[e] + pos;
  sTok[idx] = r >> 1;
  sSlot[idx] = r & 1;
  sWgt[idx] = wgt[r];
}

// ---------------- gate GEMM + gelu*gate, WMMA f16 ----------------
// block = 16-row tile (blockIdx.x) x 64 H-columns (blockIdx.y), 8 waves.
// wave w: gate g = w>>2, 16-column subtile nsub = w&3. K = FDIM (64 wmma steps).
__global__ void __launch_bounds__(256, 2)
gate_kernel(const float* __restrict__ x, const float* __restrict__ w_gating,
            const int* __restrict__ numTiles, const int* __restrict__ tileE,
            const int* __restrict__ tileR0, const int* __restrict__ tileRows,
            const int* __restrict__ sTok, _Float16* __restrict__ act) {
  if ((int)blockIdx.x >= *numTiles) return;
  __shared__ _Float16 sX[16 * 256];   // x chunk, f16
  __shared__ float sG[2 * 16 * 64];   // both gate tiles
  __shared__ int sTokS[16];
  int tid = threadIdx.x;
  int lane = tid & 31, wave = tid >> 5;
  int e = tileE[blockIdx.x];
  int r0 = tileR0[blockIdx.x];
  int rows = tileRows[blockIdx.x];
  if (tid < 16) sTokS[tid] = sTok[r0 + ((tid < rows) ? tid : 0)];

  int g = wave >> 2, nsub = wave & 3;
  int h0 = blockIdx.y * 64;
  int n = lane & 15;                  // B column within subtile
  int kb = (lane < 16) ? 0 : 16;      // B k-base (contiguous k = 16h + s, same as A)
  int m = lane & 15;                  // A row
  int ab = (lane < 16) ? 0 : 16;      // A k-base (contiguous k = 16h + s, same as B)
  // w_gating[e][g][h][f]: row h contiguous over f == B^T rows are contiguous
  const float* brow = w_gating + (((size_t)e * 2 + g) * HDIM + (size_t)(h0 + nsub * 16 + n)) * FDIM;
  v8f c = {};
  for (int ko = 0; ko < FDIM; ko += 256) {
    __syncthreads();
    for (int i = tid; i < 16 * 64; i += 256) {  // stage 16x256 fp32 -> f16
      int mr = i >> 6, c4 = i & 63;
      const float4 v4 = *(const float4*)(x + (size_t)sTokS[mr] * FDIM + ko + c4 * 4);
      _Float16* d = &sX[mr * 256 + c4 * 4];
      d[0] = (_Float16)v4.x; d[1] = (_Float16)v4.y;
      d[2] = (_Float16)v4.z; d[3] = (_Float16)v4.w;
    }
    __syncthreads();
#pragma unroll
    for (int kk = 0; kk < 256; kk += 32) {
      union { v16h v; unsigned u[8]; } a;
      const unsigned* sXu = (const unsigned*)sX;
#pragma unroll
      for (int v = 0; v < 8; ++v) {
        int koff = ab + (v << 1);  // A: 16 consecutive K per lane (+16 hi)
        a.u[v] = sXu[(m * 256 + kk + koff) >> 1];
      }
      const float* bp = brow + ko + kk + kb;   // B: 16 consecutive K per lane
      __builtin_prefetch(bp + 256, 0, 1);      // global_prefetch_b8, next K-chunk
      v16h b;
#pragma unroll
      for (int q = 0; q < 4; ++q) {
        float4 f4 = *(const float4*)(bp + q * 4);
        b[q * 4 + 0] = (_Float16)f4.x; b[q * 4 + 1] = (_Float16)f4.y;
        b[q * 4 + 2] = (_Float16)f4.z; b[q * 4 + 3] = (_Float16)f4.w;
      }
      c = __builtin_amdgcn_wmma_f32_16x16x32_f16(false, a.v, false, b, (short)0, c,
                                                 false, false);
    }
  }
  // C layout: lane n, VGPR v -> row v (+8 for hi half-wave)
#pragma unroll
  for (int v = 0; v < 8; ++v) {
    int mr = v + ((lane < 16) ? 0 : 8);
    sG[g * 1024 + mr * 64 + nsub * 16 + n] = c[v];
  }
  __syncthreads();
  for (int i = tid; i < 16 * 64; i += 256) {
    int mr = i >> 6, col = i & 63;
    if (mr < rows) {
      float g1 = sG[mr * 64 + col];
      float g2 = sG[1024 + mr * 64 + col];
      act[(size_t)(r0 + mr) * HDIM + h0 + col] = (_Float16)(gelu_tanh(g1) * g2);
    }
  }
}

// ---------------- down-proj GEMM, WMMA f16 + double-buffered async DMA ----------------
// block = 16-row tile (blockIdx.x) x 128 F-columns (blockIdx.y), 8 waves x 16 cols.
// w_linear is K-major over h -> DMA 64x128 fp32 chunks coalesced into LDS with
// global_load_async_to_lds_b128 (no VGPR round-trip), double-buffered so chunk
// i+1's DMA overlaps chunk i's WMMAs; gather B layout + cvt from LDS.
#define KC 64  // K-chunk (two 16x16x32 WMMA steps per chunk)
__global__ void __launch_bounds__(256, 2)
linear_kernel(const _Float16* __restrict__ act, const float* __restrict__ w_linear,
              const int* __restrict__ numTiles, const int* __restrict__ tileE,
              const int* __restrict__ tileR0, const int* __restrict__ tileRows,
              const int* __restrict__ sTok, const int* __restrict__ sSlot,
              const float* __restrict__ sWgt, float* __restrict__ ybuf) {
  if ((int)blockIdx.x >= *numTiles) return;
  __shared__ float sB32[2][KC * 128];   // 2 x 32KB fp32 tiles, filled by async DMA
  __shared__ int sT[16];
  __shared__ int sS[16];
  __shared__ float sW[16];
  int tid = threadIdx.x;
  int lane = tid & 31, wave = tid >> 5;
  int e = tileE[blockIdx.x];
  int r0 = tileR0[blockIdx.x];
  int rows = tileRows[blockIdx.x];
  if (tid < 16) {
    int rr = (tid < rows) ? tid : 0;
    sT[tid] = sTok[r0 + rr];
    sS[tid] = sSlot[r0 + rr];
    sW[tid] = sWgt[r0 + rr];
  }
  int f0 = blockIdx.y * 128;
  int n = lane & 15;
  int kb = (lane < 16) ? 0 : 16;
  int m = lane & 15;
  int ab = (lane < 16) ? 0 : 16;
  int fn = wave * 16 + n;
  const _Float16* arow = act + (size_t)(r0 + m) * HDIM;
  const float* wl = w_linear + (size_t)e * HDIM * FDIM + f0;
  unsigned ldsB[2] = { lds_off32(&sB32[0][0]), lds_off32(&sB32[1][0]) };

  // Issue one 64x128 fp32 chunk (8 x 16B per thread) into buffer `buf`.
  auto issue_chunk = [&](int kk, int buf) {
#pragma unroll
    for (int j = 0; j < 8; ++j) {
      int i = tid + j * 256;          // 0..2047
      int kr = i >> 5, c4 = i & 31;   // kr 0..63, c4 0..31 (x4 floats)
      async_copy_b128(ldsB[buf] + (unsigned)((kr * 128 + c4 * 4) * 4),
                      wl + (size_t)(kk + kr) * FDIM + c4 * 4);
    }
  };

  issue_chunk(0, 0);
  v8f c = {};
  for (int kk = 0, it = 0; kk < HDIM; kk += KC, ++it) {
    int cur = it & 1;
    bool more = (kk + KC) < HDIM;
    if (more) issue_chunk(kk + KC, cur ^ 1);  // overlap next DMA with compute
    if (more) wait_asynccnt<8>();             // current chunk landed (in-order)
    else      wait_asynccnt<0>();
    __syncthreads();                          // all waves' DMAs for `cur` landed
#pragma unroll
    for (int s = 0; s < 2; ++s) {
      union { v16h v; unsigned u[8]; } a;
      const unsigned* au = (const unsigned*)(arow + kk + s * 32);
#pragma unroll
      for (int v = 0; v < 8; ++v) {
        int koff = ab + (v << 1);
        a.u[v] = au[koff >> 1];
      }
      union { v16h v; _Float16 h[16]; } b;
      const float* bt = &sB32[cur][(s * 32 + kb) * 128 + fn];
#pragma unroll
      for (int q = 0; q < 8; ++q) {
        float blo = bt[(2 * q) * 128];
        float bhi = bt[(2 * q + 1) * 128];
        b.h[2 * q] = (_Float16)blo;
        b.h[2 * q + 1] = (_Float16)bhi;
      }
      c = __builtin_amdgcn_wmma_f32_16x16x32_f16(false, a.v, false, b.v, (short)0, c,
                                                 false, false);
    }
    __syncthreads();  // all waves done reading `cur` before it is re-filled
  }
#pragma unroll
  for (int v = 0; v < 8; ++v) {
    int mr = v + ((lane < 16) ? 0 : 8);
    if (mr < rows) {
      float val = c[v] * sW[mr];
      ybuf[((size_t)sT[mr] * 2 + sS[mr]) * FDIM + f0 + fn] = val;
    }
  }
}

// ---------------- combine: out = slot0 + slot1 ----------------
__global__ void combine_kernel(const float* __restrict__ ybuf, float* __restrict__ out) {
  size_t i = (size_t)blockIdx.x * blockDim.x + threadIdx.x;
  size_t N4 = (size_t)BTOK * FDIM / 4;
  if (i >= N4) return;
  size_t t = i / (FDIM / 4);
  size_t cc = i % (FDIM / 4);
  const float4* y = (const float4*)ybuf;
  float4 a = y[(t * 2 + 0) * (FDIM / 4) + cc];
  float4 b = y[(t * 2 + 1) * (FDIM / 4) + cc];
  float4 o;
  o.x = a.x + b.x; o.y = a.y + b.y; o.z = a.z + b.z; o.w = a.w + b.w;
  ((float4*)out)[i] = o;
}

extern "C" void kernel_launch(void* const* d_in, const int* in_sizes, int n_in,
                              void* d_out, int out_size, void* d_ws, size_t ws_size,
                              hipStream_t stream) {
  (void)in_sizes; (void)n_in; (void)out_size; (void)ws_size;
  const float* x   = (const float*)d_in[0];
  const float* wr  = (const float*)d_in[1];
  const float* wg  = (const float*)d_in[2];
  const float* wl  = (const float*)d_in[3];
  const float* pes = (const float*)d_in[4];
  const float* rsc = (const float*)d_in[5];
  // d_in[6] = top_k (== 2, hardcoded as TOPK)
  float* out = (float*)d_out;

  // Workspace carve (256B aligned). Total ~256 MB:
  // act f16 R*H = 128MB, ybuf f32 R*F = 128MB, metadata < 1MB.
  char* base = (char*)d_ws;
  size_t off = 0;
  auto carve = [&](size_t bytes) -> void* {
    void* p = base + off;
    off = (off + bytes + 255) & ~(size_t)255;
    return p;
  };
  int*   expertOf = (int*)  carve((size_t)RROWS * 4);
  float* wgt      = (float*)carve((size_t)RROWS * 4);
  int*   counts   = (int*)  carve(EDIM * 4);
  int*   offsets  = (int*)  carve((EDIM + 1) * 4);
  int*   cursor   = (int*)  carve(EDIM * 4);
  int*   sTok     = (int*)  carve((size_t)RROWS * 4);
  int*   sSlot    = (int*)  carve((size_t)RROWS * 4);
  float* sWgt     = (float*)carve((size_t)RROWS * 4);
  int*   tileE    = (int*)  carve(MAXTILES * 4);
  int*   tileR0   = (int*)  carve(MAXTILES * 4);
  int*   tileRows = (int*)  carve(MAXTILES * 4);
  int*   numTiles = (int*)  carve(4);
  _Float16* act   = (_Float16*)carve((size_t)RROWS * HDIM * 2);
  float* ybuf     = (float*)carve((size_t)RROWS * FDIM * 4);

  init_kernel<<<1, 32, 0, stream>>>(counts, cursor, numTiles);
  router_kernel<<<BTOK / 8, 256, 0, stream>>>(x, wr, pes, rsc, expertOf, wgt, counts);
  plan_kernel<<<1, 1, 0, stream>>>(counts, offsets, tileE, tileR0, tileRows, numTiles);
  scatter_kernel<<<RROWS / 256, 256, 0, stream>>>(expertOf, wgt, offsets, cursor,
                                                  sTok, sSlot, sWgt);
  gate_kernel<<<dim3(MAXTILES, HDIM / 64), 256, 0, stream>>>(
      x, wg, numTiles, tileE, tileR0, tileRows, sTok, act);
  linear_kernel<<<dim3(MAXTILES, FDIM / 128), 256, 0, stream>>>(
      act, wl, numTiles, tileE, tileR0, tileRows, sTok, sSlot, sWgt, ybuf);
  combine_kernel<<<(unsigned)(((size_t)BTOK * FDIM / 4 + 255) / 256), 256, 0, stream>>>(
      ybuf, out);
}
